// AttBlock_47296179863935
// MI455X (gfx1250) — matrix operand order repaired and checked by reading, not compiled, not run on hardware
//
#include <hip/hip_runtime.h>

// ---------------------------------------------------------------------------
// AttnBlock (GroupNorm -> QKV 1x1 -> softmax(QK^T/sqrt(C)) V -> out proj + x)
// N=4, C=D=512, H=W=64 (L=4096), 32 groups.
// Strategy: fp16 operands + fp32 WMMA accumulation (v_wmma_f32_16x16x32_f16),
// flash-style attention (no 268MB score tensor), all intermediates in d_ws.
// Workspace requirement: 4*512KB (weights f16) + 5 * 16MB (h,q,k,v,o f16)
//                      = 85,983,232 bytes.
// ---------------------------------------------------------------------------

typedef _Float16 half_t;
typedef __attribute__((ext_vector_type(16))) _Float16 v16h;
typedef __attribute__((ext_vector_type(8)))  _Float16 v8h;
typedef __attribute__((ext_vector_type(8)))  float    v8f;

#define C_DIM 512
#define D_DIM 512
#define L_DIM 4096
#define NB 4
#define EPS_GN 1e-5f

__device__ __forceinline__ int lane_id() { return threadIdx.x & 31; }

__device__ __forceinline__ v8f vzero8() {
  v8f r;
#pragma unroll
  for (int i = 0; i < 8; ++i) r[i] = 0.f;
  return r;
}

__device__ __forceinline__ v8f wmma_f16(v16h a, v16h b, v8f c) {
  // (neg_a, A, neg_b, B, c_mod, C, reuse_a, reuse_b)
  return __builtin_amdgcn_wmma_f32_16x16x32_f16(false, a, false, b, (short)0, c,
                                                false, false);
}

// A-fragment (16x32 f16, M x K): lanes 0-15 -> row=lane, K {0..7,16..23};
// lanes 16-31 -> row=lane-16, K {8..15,24..31}. Source: row-major [rows][stride].
__device__ __forceinline__ v16h frag_a(const half_t* base, int row0, int stride,
                                       int k0) {
  int l = lane_id();
  const half_t* p =
      base + (size_t)(row0 + (l & 15)) * stride + k0 + ((l & 16) ? 8 : 0);
  v8h lo = *(const v8h*)p;
  v8h hh = *(const v8h*)(p + 16);
  return __builtin_shufflevector(lo, hh, 0, 1, 2, 3, 4, 5, 6, 7, 8, 9, 10, 11,
                                 12, 13, 14, 15);
}

// B-fragment (32x16 f16, K x N): lane n holds col N=n%16, K {0..7,16..23} for
// n<16, K {8..15,24..31} for n>=16.
// Sourced from row-major [n][K] (i.e. W[n][k] with k contiguous).
__device__ __forceinline__ v16h frag_b(const half_t* base, int n0, int stride,
                                       int k0) {
  int l = lane_id();
  const half_t* p =
      base + (size_t)(n0 + (l & 15)) * stride + k0 + ((l & 16) ? 8 : 0);
  v8h lo = *(const v8h*)p;
  v8h hh = *(const v8h*)(p + 16);
  return __builtin_shufflevector(lo, hh, 0, 1, 2, 3, 4, 5, 6, 7, 8, 9, 10, 11,
                                 12, 13, 14, 15);
}

// ---------------------------------------------------------------------------
// fp32 -> fp16 convert (weights)
// ---------------------------------------------------------------------------
__global__ void f32_to_f16_kernel(const float* __restrict__ in,
                                  half_t* __restrict__ out, int n) {
  int i = blockIdx.x * 256 + threadIdx.x;
  if (i < n) out[i] = (half_t)in[i];
}

// ---------------------------------------------------------------------------
// GroupNorm: x[N][C][L] fp32 -> h_t[N][L][C] fp16 (transposed for GEMM A/B use)
// One block per (group, batch): 16 channels x 4096 positions.
// ---------------------------------------------------------------------------
__global__ __launch_bounds__(256) void groupnorm_h16(
    const float* __restrict__ x, const float* __restrict__ gsc,
    const float* __restrict__ gbi, half_t* __restrict__ h_t) {
  int n = blockIdx.y, g = blockIdx.x;
  int cbase = g * 16;
  const float* xp = x + ((size_t)n * C_DIM + cbase) * L_DIM;  // 16*L contiguous
  int tid = threadIdx.x;

  float s = 0.f, ss = 0.f;
  for (int i = tid; i < 16 * L_DIM; i += 256) {
    float v = xp[i];
    s += v;
    ss += v * v;
  }
  __shared__ float rs[256], rss[256];
  rs[tid] = s;
  rss[tid] = ss;
  __syncthreads();
  for (int o = 128; o > 0; o >>= 1) {
    if (tid < o) {
      rs[tid] += rs[tid + o];
      rss[tid] += rss[tid + o];
    }
    __syncthreads();
  }
  __shared__ float smean, srstd;
  if (tid == 0) {
    float mean = rs[0] * (1.f / 65536.f);
    float var = rss[0] * (1.f / 65536.f) - mean * mean;
    smean = mean;
    srstd = rsqrtf(var + EPS_GN);
  }
  __syncthreads();
  float mean = smean, rstd = srstd;

  // write h_t[n][l][cbase+c]; 16 consecutive threads write 32B contiguous
  for (int i = tid; i < 16 * L_DIM; i += 256) {
    int c = i & 15, ll = i >> 4;
    float y = (xp[(size_t)c * L_DIM + ll] - mean) * rstd * gsc[cbase + c] +
              gbi[cbase + c];
    h_t[((size_t)n * L_DIM + ll) * C_DIM + cbase + c] = (half_t)y;
  }
}

// ---------------------------------------------------------------------------
// Generic WMMA GEMM: D[m][n] = sum_k A[m][k] * B[n][k] (+bias)(+residual)
// A: f16 [M][K], B: f16 [Nn][K], out row-major [M][Nn] (f16 or f32).
// Each wave: 16m x 64n tile; 128-thread blocks = 4 waves.
// ---------------------------------------------------------------------------
template <bool BIAS_PER_N, bool OUT_F16, bool RESID>
__global__ __launch_bounds__(128) void gemm_rr(
    const half_t* __restrict__ A, size_t sbA, const half_t* __restrict__ B,
    size_t sbB, void* __restrict__ Out, size_t sbO,
    const float* __restrict__ bias, const float* __restrict__ resid, size_t sbR,
    int M, int Nn, int K) {
  int b = blockIdx.y;
  int wid = blockIdx.x * 4 + (threadIdx.x >> 5);
  int tilesM = M >> 4, tilesN = Nn >> 6;
  if (wid >= tilesM * tilesN) return;  // wave-uniform
  int m0 = (wid % tilesM) << 4;
  int n0 = (wid / tilesM) << 6;
  const half_t* Ab = A + (size_t)b * sbA;
  const half_t* Bb = B + (size_t)b * sbB;

  v8f acc[4];
#pragma unroll
  for (int j = 0; j < 4; ++j) acc[j] = vzero8();

  for (int k0 = 0; k0 < K; k0 += 32) {
    v16h a = frag_a(Ab, m0, K, k0);
#pragma unroll
    for (int j = 0; j < 4; ++j) {
      v16h bf = frag_b(Bb, n0 + j * 16, K, k0);
      acc[j] = wmma_f16(a, bf, acc[j]);
    }
  }

  int l = lane_id();
  int col = l & 15;
  int rb = (l & 16) ? 8 : 0;
#pragma unroll
  for (int j = 0; j < 4; ++j) {
    int n = n0 + j * 16 + col;
    float bn = BIAS_PER_N ? bias[n] : 0.f;
#pragma unroll
    for (int r = 0; r < 8; ++r) {
      int m = m0 + rb + r;
      float val = acc[j][r] + (BIAS_PER_N ? bn : bias[m]);
      size_t idx = (size_t)m * Nn + n;
      if (RESID) val += resid[(size_t)b * sbR + idx];
      if (OUT_F16)
        ((half_t*)Out)[(size_t)b * sbO + idx] = (half_t)val;
      else
        ((float*)Out)[(size_t)b * sbO + idx] = val;
    }
  }
}

// ---------------------------------------------------------------------------
// Flash attention. q_t,k_t: [N][L][D] f16; v: [N][D][L] f16; o_t: [N][L][D].
// Each wave: 16 queries, full D=512 output (32 v8f accumulators).
// S^T tiles (j x i) are computed so the probability C-fragment is *exactly*
// the A-fragment layout of P for att@V^T -- no LDS shuffle of P needed.
// Row reductions use a single lane<->lane+16 swap (wave32).
// ---------------------------------------------------------------------------
__global__ __launch_bounds__(128, 1) void attn_flash(
    const half_t* __restrict__ q_t, const half_t* __restrict__ k_t,
    const half_t* __restrict__ v_dl, half_t* __restrict__ o_t) {
  const int D = D_DIM, L = L_DIM;
  __shared__ __align__(32) half_t qs[4][16 * D_DIM];
  int b = blockIdx.y;
  int w = threadIdx.x >> 5;
  int wid = blockIdx.x * 4 + w;
  int i0 = wid << 4;
  const half_t* qb = q_t + ((size_t)b * L + i0) * D;
  const half_t* kb = k_t + (size_t)b * L * D;
  const half_t* vb = v_dl + (size_t)b * D * L;
  half_t* ob = o_t + (size_t)b * L * D;
  int l = lane_id();
  int col = l & 15;
  int rb = (l & 16) ? 8 : 0;
  int srcoff = (l & 16) >> 1;  // 0 for lanes<16, 8 for lanes>=16

  // stage this wave's 16 query rows into LDS (one flat 16KB copy)
  {
    const v8h* src = (const v8h*)qb;
    v8h* dst = (v8h*)qs[w];
    for (int idx = l; idx < (16 * D_DIM) / 8; idx += 32) dst[idx] = src[idx];
  }

  v8f O[32];
#pragma unroll
  for (int t = 0; t < 32; ++t) O[t] = vzero8();
  float m_run = -3.0e38f, l_run = 0.f;
  const float scale = 0.044194173824159216f;  // 512^-0.5

  for (int j0 = 0; j0 < L; j0 += 32) {
    // S^T tiles: rows j (keys), cols i (queries). K-reduction over D=512.
    v8f s0 = vzero8(), s1 = vzero8();
#pragma unroll
    for (int k0 = 0; k0 < D_DIM; k0 += 32) {
      v16h bq = frag_b(qs[w], 0, D_DIM, k0);  // LDS reads (ds/flat)
      v16h a0 = frag_a(kb, j0, D, k0);
      v16h a1 = frag_a(kb, j0 + 16, D, k0);
      s0 = wmma_f16(a0, bq, s0);
      s1 = wmma_f16(a1, bq, s1);
    }

    // lane holds 16 scores for query i = i0 + col; partner lane (xor 16)
    // holds the other 16 j's of this 32-key chunk.
    float sl[16];
#pragma unroll
    for (int r = 0; r < 8; ++r) {
      sl[r] = s0[r] * scale;
      sl[8 + r] = s1[r] * scale;
    }
    float mx = sl[0];
#pragma unroll
    for (int e = 1; e < 16; ++e) mx = fmaxf(mx, sl[e]);
    mx = fmaxf(mx, __shfl_xor(mx, 16, 32));
    float m_new = fmaxf(m_run, mx);
    float alpha = __expf(m_run - m_new);

    v16h pa;  // P as A-fragment (16i x 32j) -- direct from S^T C-fragments
    float lsum = 0.f;
#pragma unroll
    for (int e = 0; e < 16; ++e) {
      float p = __expf(sl[e] - m_new);
      lsum += p;
      pa[e] = (half_t)p;
    }
    lsum += __shfl_xor(lsum, 16, 32);
    l_run = l_run * alpha + lsum;
    m_run = m_new;

    // broadcast per-row alpha into the O-fragment row layout (M = r + rb)
    float ar[8];
#pragma unroll
    for (int r = 0; r < 8; ++r) ar[r] = __shfl(alpha, r + srcoff, 32);

#pragma unroll
    for (int t = 0; t < 32; ++t) {
#pragma unroll
      for (int r = 0; r < 8; ++r) O[t][r] *= ar[r];
      v16h bv = frag_b(vb, t * 16, L, j0);  // V^T fragment: contiguous j
      O[t] = wmma_f16(pa, bv, O[t]);
    }
  }

  float li[8];
#pragma unroll
  for (int r = 0; r < 8; ++r) {
    float lr = __shfl(l_run, r + srcoff, 32);
    li[r] = 1.f / lr;
  }
#pragma unroll
  for (int t = 0; t < 32; ++t) {
#pragma unroll
    for (int r = 0; r < 8; ++r)
      ob[(size_t)(i0 + rb + r) * D + t * 16 + col] = (half_t)(O[t][r] * li[r]);
  }
}

// ---------------------------------------------------------------------------
extern "C" void kernel_launch(void* const* d_in, const int* in_sizes, int n_in,
                              void* d_out, int out_size, void* d_ws,
                              size_t ws_size, hipStream_t stream) {
  const float* x = (const float*)d_in[0];
  const float* gsc = (const float*)d_in[1];
  const float* gbi = (const float*)d_in[2];
  const float* wq = (const float*)d_in[3];
  const float* bq = (const float*)d_in[4];
  const float* wk = (const float*)d_in[5];
  const float* bk = (const float*)d_in[6];
  const float* wv = (const float*)d_in[7];
  const float* bv = (const float*)d_in[8];
  const float* wo = (const float*)d_in[9];
  const float* bo = (const float*)d_in[10];
  float* out = (float*)d_out;

  const int M_L = L_DIM, Nd = D_DIM, K = C_DIM;
  const size_t WEL = (size_t)C_DIM * D_DIM;  // 262144 weight elements
  const size_t TEL = (size_t)L_DIM * D_DIM;  // 2097152 per-batch tensor elems

  half_t* wq_h = (half_t*)d_ws;
  half_t* wk_h = wq_h + WEL;
  half_t* wv_h = wk_h + WEL;
  half_t* wo_h = wv_h + WEL;
  half_t* h_t = wo_h + WEL;              // [N][L][C]
  half_t* q_t = h_t + (size_t)NB * TEL;  // [N][L][D]
  half_t* k_t = q_t + (size_t)NB * TEL;  // [N][L][D]
  half_t* v_dl = k_t + (size_t)NB * TEL; // [N][D][L]
  half_t* o_t = v_dl + (size_t)NB * TEL; // [N][L][D]

  // 1) weights -> f16
  dim3 cg((unsigned)((WEL + 255) / 256));
  f32_to_f16_kernel<<<cg, 256, 0, stream>>>(wq, wq_h, (int)WEL);
  f32_to_f16_kernel<<<cg, 256, 0, stream>>>(wk, wk_h, (int)WEL);
  f32_to_f16_kernel<<<cg, 256, 0, stream>>>(wv, wv_h, (int)WEL);
  f32_to_f16_kernel<<<cg, 256, 0, stream>>>(wo, wo_h, (int)WEL);

  // 2) GroupNorm -> h_t[N][L][C] f16
  groupnorm_h16<<<dim3(32, NB), 256, 0, stream>>>(x, gsc, gbi, h_t);

  // 3) q_t[l][d] / k_t[l][d]: A = h_t (batched), B = W rows, bias per-N
  {
    int waves = (M_L / 16) * (Nd / 64);
    dim3 grid((unsigned)((waves + 3) / 4), NB);
    gemm_rr<true, true, false><<<grid, 128, 0, stream>>>(
        h_t, TEL, wq_h, 0, q_t, TEL, bq, nullptr, 0, M_L, Nd, K);
    gemm_rr<true, true, false><<<grid, 128, 0, stream>>>(
        h_t, TEL, wk_h, 0, k_t, TEL, bk, nullptr, 0, M_L, Nd, K);
  }
  // 4) v[d][l]: A = wv rows (shared), B = h_t (batched), bias per-M
  {
    int waves = (Nd / 16) * (M_L / 64);
    dim3 grid((unsigned)((waves + 3) / 4), NB);
    gemm_rr<false, true, false><<<grid, 128, 0, stream>>>(
        wv_h, 0, h_t, TEL, v_dl, TEL, bv, nullptr, 0, Nd, M_L, K);
  }

  // 5) flash attention -> o_t[N][L][D]
  attn_flash<<<dim3(L_DIM / 64, NB), 128, 0, stream>>>(q_t, k_t, v_dl, o_t);

  // 6) out[d'][l] = wo @ o + bo + x (residual), f32 output
  {
    int waves = (Nd / 16) * (M_L / 64);
    dim3 grid((unsigned)((waves + 3) / 4), NB);
    gemm_rr<false, false, true><<<grid, 128, 0, stream>>>(
        wo_h, 0, o_t, TEL, out, (size_t)D_DIM * L_DIM, bo, x,
        (size_t)C_DIM * L_DIM, Nd, M_L, D_DIM);
  }
}
